// MultiHeadAttention_27470610825958
// MI455X (gfx1250) — matrix operand order repaired and checked by reading, not compiled, not run on hardware
//
#include <hip/hip_runtime.h>
#include <stdint.h>

typedef __attribute__((ext_vector_type(16))) _Float16 v16h;
typedef __attribute__((ext_vector_type(8)))  float    v8f;

#define BATCH  4
#define SEQ    1024
#define DMODEL 1024
#define NHEAD  16
#define DKH    64   // head dim

static __device__ __forceinline__ v8f wmma_f16(v16h a, v16h b, v8f c) {
  return __builtin_amdgcn_wmma_f32_16x16x32_f16(false, a, false, b,
                                                (short)0, c, false, false);
}

// ---------------------------------------------------------------------------
// elementwise f32 -> f16
// ---------------------------------------------------------------------------
__global__ __launch_bounds__(256) void f32_to_f16_kernel(
    const float* __restrict__ src, _Float16* __restrict__ dst, int n) {
  int i = blockIdx.x * 256 + threadIdx.x;
  if (i < n) dst[i] = (_Float16)src[i];
}

// ---------------------------------------------------------------------------
// WMMA GEMM: C[M,N] = A[M,K] * B[K,N], f16 in, f32 accum.
// Block tile 64x128, 256 threads = 8 waves in 2x4 grid, wave tile 32x32
// (4 WMMAs per k-step sharing 2 A-frags + 2 B-frags).
// A staged in fragment-major LDS order: 16B global read -> 16B LDS write ->
// 32B per-lane ds_load_b128 fragment reads.
// OUT_MODE 0: f32 flat to Cf          (final projection -> d_out)
// OUT_MODE 1: f16 [B,H,S,DK] to Ch    (Q, K)
// OUT_MODE 2: f16 [B,H,DK,S] to Ch    (V transposed, for contiguous P*V loads)
// ---------------------------------------------------------------------------
template <int OUT_MODE>
__global__ __launch_bounds__(256) void gemm64x128_wmma(
    const _Float16* __restrict__ A, const _Float16* __restrict__ Bw,
    _Float16* __restrict__ Ch, float* __restrict__ Cf,
    int M, int N, int K) {
  __shared__ _Float16 Af[4][32][16];   // 4 KB: [m-subtile][lane][frag elem]
  __shared__ _Float16 Bf[8][32][16];   // 8 KB: [n-subtile][lane][frag elem]

  const int tid  = threadIdx.x;
  const int lane = tid & 31;
  const int wave = tid >> 5;
  const int hlf  = lane >> 4;
  const int lr   = lane & 15;

  const int m0 = blockIdx.y * 64;
  const int n0 = blockIdx.x * 128;
  const int wy = wave >> 2;   // 0..1 -> rows [wy*32,+32)
  const int wx = wave & 3;    // 0..3 -> cols [wx*32,+32)

  // A staging map: thread -> one 8-half chunk (row ar, halves acseg*8..+7)
  const int ar    = tid >> 2;            // 0..63
  const int acseg = tid & 3;             // 0..3
  const int alane = (acseg & 1) * 16 + (ar & 15);
  const int aib   = (acseg >> 1) * 8;
  const int asub  = ar >> 4;

  v8f acc[2][2] = {{{}, {}}, {{}, {}}};

  for (int k0 = 0; k0 < K; k0 += 32) {
    // ---- stage A tile 64x32 (fragment-major): one b128 in, one b128 out ----
    {
      uint4 d = *(const uint4*)(A + (size_t)(m0 + ar) * K + k0 + acseg * 8);
      *(uint4*)&Af[asub][alane][aib] = d;
    }
    // ---- stage B tile 32x128: coalesced b128 reads, swizzled b16 writes ----
#pragma unroll
    for (int cc = 0; cc < 2; ++cc) {
      int ch = tid + cc * 256;           // 0..511
      int bk = ch >> 4, cseg = ch & 15;  // row 0..31, 8-half col segment
      union { uint4 v; _Float16 h[8]; } cv;
      cv.v = *(const uint4*)(Bw + (size_t)(k0 + bk) * N + n0 + cseg * 8);
#pragma unroll
      for (int j = 0; j < 8; ++j) {
        int c = cseg * 8 + j;
        Bf[c >> 4][((bk >> 3) & 1) * 16 + (c & 15)][((bk >> 4) << 3) + (bk & 7)] = cv.h[j];
      }
    }
    if (k0 + 32 < K) {  // global_prefetch_b8 next k-tile
      __builtin_prefetch(A + (size_t)(m0 + ar) * K + (k0 + 32) + acseg * 8, 0, 1);
      __builtin_prefetch(Bw + (size_t)(k0 + 32 + (tid >> 4)) * N + n0 + (tid & 15) * 8, 0, 1);
    }
    __syncthreads();

    v16h a0 = *(const v16h*)&Af[2 * wy + 0][lane][0];
    v16h a1 = *(const v16h*)&Af[2 * wy + 1][lane][0];
    v16h b0 = *(const v16h*)&Bf[2 * wx + 0][lane][0];
    v16h b1 = *(const v16h*)&Bf[2 * wx + 1][lane][0];
    acc[0][0] = wmma_f16(a0, b0, acc[0][0]);
    acc[0][1] = wmma_f16(a0, b1, acc[0][1]);
    acc[1][0] = wmma_f16(a1, b0, acc[1][0]);
    acc[1][1] = wmma_f16(a1, b1, acc[1][1]);
    __syncthreads();
  }

  // store: D row = r + 8*hlf, col = lr
#pragma unroll
  for (int su = 0; su < 2; ++su)
#pragma unroll
    for (int nu = 0; nu < 2; ++nu)
#pragma unroll
      for (int r = 0; r < 8; ++r) {
        int gm = m0 + (2 * wy + su) * 16 + hlf * 8 + r;
        int gn = n0 + (2 * wx + nu) * 16 + lr;
        float v = acc[su][nu][r];
        if (OUT_MODE == 0) {
          Cf[(size_t)gm * N + gn] = v;
        } else {
          int b = gm >> 10, s = gm & 1023;
          int h = gn >> 6,  dk = gn & 63;
          if (OUT_MODE == 1)
            Ch[(((size_t)(b * NHEAD + h)) * SEQ + s) * DKH + dk] = (_Float16)v;
          else
            Ch[(((size_t)(b * NHEAD + h)) * DKH + dk) * SEQ + s] = (_Float16)v;
        }
      }
}

// ---------------------------------------------------------------------------
// Fused attention: one block per (b, h, 16 query rows). 256 threads, 8 waves.
//   phase 1: scores = Q*K^T/8 + bias -> f16 LDS [16][1024]
//            (Q fragments hoisted; K fragments contiguous 32B v16h loads)
//   phase 2: softmax via wave shfl reductions; write attn f32 (coalesced)
//   phase 3: ctx = P*V with V stored [B,H,DK,S] -> contiguous B fragments;
//            ds_add_f32 cross-wave reduction
// ---------------------------------------------------------------------------
__global__ __launch_bounds__(256) void attn_kernel(
    const _Float16* __restrict__ Q, const _Float16* __restrict__ Km,
    const _Float16* __restrict__ Vt, const float* __restrict__ bias,
    float* __restrict__ attnOut, _Float16* __restrict__ ctxOut) {
  __shared__ _Float16 sc[16][SEQ + 8];   // 33 KB scores/probs
  __shared__ _Float16 qf[2][32][16];     //  2 KB Q tile, fragment-major
  __shared__ float ctxAcc[16][DKH];      //  4 KB ctx reduction

  const int qt = blockIdx.x, h = blockIdx.y, b = blockIdx.z;
  const int q0 = qt * 16;
  const size_t bh = (size_t)(b * NHEAD + h);
  const _Float16* Qbh  = Q  + bh * SEQ * DKH;
  const _Float16* Kbh  = Km + bh * SEQ * DKH;
  const _Float16* Vtbh = Vt + bh * DKH * SEQ;   // [DK][S]

  const int tid  = threadIdx.x;
  const int lane = tid & 31;
  const int wave = tid >> 5;
  const int hlf  = lane >> 4;
  const int lr   = lane & 15;

  // load Q tile 16x64 straight into fragment-major order (b128 in/out)
  if (tid < 128) {
    int r = tid >> 3, cseg = tid & 7;          // row, 8-half segment
    uint4 d = *(const uint4*)(Qbh + (size_t)(q0 + r) * DKH + cseg * 8);
    int step = cseg >> 2, c2 = cseg & 3;
    *(uint4*)&qf[step][(c2 >> 1) * 16 + r][(c2 & 1) * 8] = d;
  }
  for (int i = tid; i < 16 * DKH; i += 256) ((float*)ctxAcc)[i] = 0.0f;
  __syncthreads();

  // ---- phase 1: scores; wave w covers keys [w*128, +128) ----
  const float scale = 0.125f;  // 1/sqrt(64)
  v16h aq0 = *(const v16h*)&qf[0][lane][0];
  v16h aq1 = *(const v16h*)&qf[1][lane][0];
  for (int t = 0; t < 8; ++t) {
    int kb = wave * 128 + t * 16;
    const _Float16* krow = Kbh + (size_t)(kb + lr) * DKH + hlf * 16;
    if (t < 7)
      __builtin_prefetch(krow + 16 * DKH, 0, 1);
    v16h bb0 = *(const v16h*)(krow);
    v16h bb1 = *(const v16h*)(krow + 32);
    v8f acc = {};
    acc = wmma_f16(aq0, bb0, acc);
    acc = wmma_f16(aq1, bb1, acc);
#pragma unroll
    for (int r = 0; r < 8; ++r) {
      int qr  = hlf * 8 + r;
      int key = kb + lr;
      float s = acc[r] * scale +
                bias[((size_t)h * SEQ + (q0 + qr)) * SEQ + key];
      sc[qr][key] = (_Float16)s;
    }
  }
  __syncthreads();

  // ---- phase 2: softmax; each wave owns 2 rows ----
  float* attnBase = attnOut + (bh * SEQ + q0) * SEQ;
#pragma unroll
  for (int rr = 0; rr < 2; ++rr) {
    int row = wave * 2 + rr;
    float m = -1e30f;
    for (int j = 0; j < 32; ++j)
      m = fmaxf(m, (float)sc[row][lane + j * 32]);
#pragma unroll
    for (int off = 16; off > 0; off >>= 1) m = fmaxf(m, __shfl_xor(m, off, 32));
    float sum = 0.0f;
    for (int j = 0; j < 32; ++j)
      sum += __expf((float)sc[row][lane + j * 32] - m);
#pragma unroll
    for (int off = 16; off > 0; off >>= 1) sum += __shfl_xor(sum, off, 32);
    float inv = 1.0f / sum;
    for (int j = 0; j < 32; ++j) {
      int col = lane + j * 32;
      float p = __expf((float)sc[row][col] - m) * inv;
      sc[row][col] = (_Float16)p;
      attnBase[(size_t)row * SEQ + col] = p;  // coalesced across lanes
    }
  }
  __syncthreads();

  // ---- phase 3: ctx = P*V; wave w contracts keys [w*128, +128) ----
  {
    v8f acc[4] = {};
    for (int step = 0; step < 4; ++step) {
      int kbase = wave * 128 + step * 32;
      v16h a;
#pragma unroll
      for (int i = 0; i < 16; ++i) {
        int kk = kbase + hlf * 16 + i;
        a[i] = sc[lr][kk];
      }
#pragma unroll
      for (int nt = 0; nt < 4; ++nt) {
        // V^T row (dk = nt*16+lr): 16 contiguous halves at key kbase+hlf*16
        v16h bb = *(const v16h*)(Vtbh + (size_t)(nt * 16 + lr) * SEQ +
                                 kbase + hlf * 16);
        acc[nt] = wmma_f16(a, bb, acc[nt]);
      }
    }
#pragma unroll
    for (int nt = 0; nt < 4; ++nt)
#pragma unroll
      for (int r = 0; r < 8; ++r)
        atomicAdd(&ctxAcc[hlf * 8 + r][nt * 16 + lr], acc[nt][r]);  // ds_add_f32
  }
  __syncthreads();

  // write ctx tile as f16 in flat [B,S,D] layout for the final projection
  _Float16* ctxBase = ctxOut + ((size_t)b * SEQ + q0) * DMODEL + h * DKH;
  for (int i = tid; i < 16 * DKH; i += 256) {
    int r = i >> 6, c = i & 63;
    ctxBase[(size_t)r * DMODEL + c] = (_Float16)ctxAcc[r][c];
  }
}

// ---------------------------------------------------------------------------
extern "C" void kernel_launch(void* const* d_in, const int* in_sizes, int n_in,
                              void* d_out, int out_size, void* d_ws,
                              size_t ws_size, hipStream_t stream) {
  (void)in_sizes; (void)n_in; (void)out_size; (void)ws_size;
  const float* x    = (const float*)d_in[0];
  const float* bias = (const float*)d_in[1];
  const float* Wq   = (const float*)d_in[2];
  const float* Wk   = (const float*)d_in[3];
  const float* Wv   = (const float*)d_in[4];
  const float* Wo   = (const float*)d_in[5];

  float* out  = (float*)d_out;                               // [4,1024,1024]
  float* attn = out + (size_t)BATCH * SEQ * DMODEL;          // [4,16,1024,1024]

  char* ws = (char*)d_ws;
  _Float16* xh  = (_Float16*)(ws);                           // 8 MB
  _Float16* wqh = (_Float16*)(ws + ((size_t)8  << 20));      // 2 MB
  _Float16* wkh = (_Float16*)(ws + ((size_t)10 << 20));
  _Float16* wvh = (_Float16*)(ws + ((size_t)12 << 20));
  _Float16* woh = (_Float16*)(ws + ((size_t)14 << 20));
  _Float16* Qh  = (_Float16*)(ws + ((size_t)16 << 20));      // 8 MB each
  _Float16* Kh  = (_Float16*)(ws + ((size_t)24 << 20));
  _Float16* Vth = (_Float16*)(ws + ((size_t)32 << 20));      // V transposed
  _Float16* ctx = (_Float16*)(ws + ((size_t)40 << 20));      // 8 MB

  const int M  = BATCH * SEQ;        // 4096
  const int nx = M * DMODEL;         // 4M
  const int nw = DMODEL * DMODEL;    // 1M

  f32_to_f16_kernel<<<(nx + 255) / 256, 256, 0, stream>>>(x,  xh,  nx);
  f32_to_f16_kernel<<<(nw + 255) / 256, 256, 0, stream>>>(Wq, wqh, nw);
  f32_to_f16_kernel<<<(nw + 255) / 256, 256, 0, stream>>>(Wk, wkh, nw);
  f32_to_f16_kernel<<<(nw + 255) / 256, 256, 0, stream>>>(Wv, wvh, nw);
  f32_to_f16_kernel<<<(nw + 255) / 256, 256, 0, stream>>>(Wo, woh, nw);

  dim3 gGemm(DMODEL / 128, M / 64);  // (8, 64)
  gemm64x128_wmma<1><<<gGemm, 256, 0, stream>>>(xh, wqh, Qh,  nullptr, M, DMODEL, DMODEL);
  gemm64x128_wmma<1><<<gGemm, 256, 0, stream>>>(xh, wkh, Kh,  nullptr, M, DMODEL, DMODEL);
  gemm64x128_wmma<2><<<gGemm, 256, 0, stream>>>(xh, wvh, Vth, nullptr, M, DMODEL, DMODEL);

  dim3 gAttn(SEQ / 16, NHEAD, BATCH);
  attn_kernel<<<gAttn, 256, 0, stream>>>(Qh, Kh, Vth, bias, attn, ctx);

  gemm64x128_wmma<0><<<gGemm, 256, 0, stream>>>(ctx, woh, nullptr, out, M, DMODEL, DMODEL);
}
